// Former_33217277067492
// MI455X (gfx1250) — matrix operand order repaired and checked by reading, not compiled, not run on hardware
//
#include <hip/hip_runtime.h>

// MI455X / gfx1250 implementation of "Former".
// - All 1x1 convs are bf16 WMMA GEMMs (v_wmma_f32_16x16x32_bf16, fp32 accum).
// - Activations are bf16 NHWC in workspace. All tile-granularity global traffic
//   (A tile in, residual tile in, C tile out) goes through the gfx1250 async
//   LDS engine (global_load_async_to_lds_b128 / global_store_async_from_lds_b128,
//   ASYNCcnt). W is staged transposed so B fragments are 2x ds_load_b128.
// - K and N are compile-time: all strides fold into instruction immediates.
// - Local 3x3 attention is an LDS-halo stencil kernel (per batch*head), f32 math.
// - Workspace: 5 rotating slots of 131072*128 bf16 (160 MiB total).

typedef __bf16 bf16_t;
typedef __attribute__((ext_vector_type(16))) __bf16 v16bf;
typedef __attribute__((ext_vector_type(8)))  __bf16 v8bf;
typedef __attribute__((ext_vector_type(8)))  float  v8f;

#define RSQ1PE 0.9999950000374997f  // 1/sqrt(1 + 1e-5)  (BN with running_var=1)
#define NEG_SLOPE 0.1f
#define USE_ASYNC_COPY 1

#if USE_ASYNC_COPY && defined(__gfx1250__)
#define ASYNC_PATH 1
#else
#define ASYNC_PATH 0
#endif

// ---- async LDS <-> global helpers (16-byte chunks) -------------------------
__device__ __forceinline__ void async_ld_b128(uint32_t lds_off, const void* gaddr) {
#if ASYNC_PATH
  asm volatile("global_load_async_to_lds_b128 %0, %1, off"
               :: "v"(lds_off), "v"((uint64_t)(uintptr_t)gaddr) : "memory");
#else
  (void)lds_off; (void)gaddr;
#endif
}
__device__ __forceinline__ void async_st_b128(const void* gaddr, uint32_t lds_off) {
#if ASYNC_PATH
  asm volatile("global_store_async_from_lds_b128 %0, %1, off"
               :: "v"((uint64_t)(uintptr_t)gaddr), "v"(lds_off) : "memory");
#else
  (void)gaddr; (void)lds_off;
#endif
}
__device__ __forceinline__ void async_wait0() {
#if ASYNC_PATH
  asm volatile("s_wait_asynccnt 0" ::: "memory");
#endif
}
__device__ __forceinline__ uint32_t lds_off_of(const void* p) {
  return (uint32_t)(uintptr_t)p;  // low 32 bits of a generic LDS pointer = LDS offset
}

// ---------------------------------------------------------------- conv0 ----
// y = leaky_relu(bn(x @ W0 + b0)), x NCHW (2,3,256,256) f32 -> out NHWC bf16
__global__ void conv0_kernel(const float* __restrict__ x, const float* __restrict__ w,
                             const float* __restrict__ bias, const float* __restrict__ g,
                             const float* __restrict__ bt, bf16_t* __restrict__ out) {
  int p  = blockIdx.x * blockDim.x + threadIdx.x;  // 0..131071
  int b  = p >> 16;
  int sp = p & 0xFFFF;
  const float* xb = x + (size_t)b * 3 * 65536 + sp;
  float x0 = xb[0], x1 = xb[65536], x2 = xb[131072];
  bf16_t* op = out + (size_t)p * 32;
#pragma unroll
  for (int c = 0; c < 32; ++c) {
    float v = fmaf(x0, w[c], fmaf(x1, w[32 + c], fmaf(x2, w[64 + c], bias[c])));
    v = v * (g[c] * RSQ1PE) + bt[c];
    op[c] = (bf16_t)(v > 0.f ? v : NEG_SLOPE * v);
  }
}

// ----------------------------------------------------------- WMMA GEMM ----
// out[M,N] = epilogue(A[M,K] @ W[K,N]); M = 131072; K,N compile-time.
// MODE 0: none (qkv).  MODE 1: +bias (w1).
// MODE 2: leaky(bn2(yres + relu(bn1(A@W))))  (wcat + residual tail, fused).
// Block = 256 threads (8 waves): 64 pixel-rows per block, waves 4(M) x 2(N).
template <int MODE, int K, int N>
__global__ void __launch_bounds__(256) gemm_kernel(
    const bf16_t* __restrict__ A, const float* __restrict__ W,
    const float* __restrict__ bias, const bf16_t* __restrict__ yres,
    const float* __restrict__ g1, const float* __restrict__ bt1,
    const float* __restrict__ g2, const float* __restrict__ bt2,
    bf16_t* __restrict__ out) {
  constexpr int KT  = K / 32;   // k-steps
  constexpr int KP  = K + 8;    // padded transposed-W stride (bank spread)
  constexpr int NT2 = N / 32;   // n-tiles per wave (2 waves along N)
  constexpr int NSH = (N == 32) ? 5 : (N == 64) ? 6 : 7;
  __shared__ __attribute__((aligned(16))) bf16_t lA[64 * K];   // [64][K] row-major
  __shared__ __attribute__((aligned(16))) bf16_t lWt[N * KP];  // [N][KP] (W^T, padded)
  __shared__ __attribute__((aligned(16))) bf16_t lOut[64 * N]; // C tile (and yres in MODE2)

  const int tid = threadIdx.x;
  const int m0  = blockIdx.x * 64;

  // ---- Stage A (and yres for MODE2): contiguous bf16 tiles -> async to LDS.
  const bf16_t* Ab = A + (size_t)m0 * K;
#if ASYNC_PATH
  {
    const uint32_t la = lds_off_of(lA);
    for (int t = tid; t < 64 * K / 8; t += 256)
      async_ld_b128(la + (uint32_t)t * 16u, Ab + t * 8);
    if (MODE == 2) {
      const bf16_t* Yb = yres + (size_t)m0 * N;
      const uint32_t lo = lds_off_of(lOut);
      for (int t = tid; t < 64 * N / 8; t += 256)
        async_ld_b128(lo + (uint32_t)t * 16u, Yb + t * 8);
    }
  }
#else
  {
    const uint32_t* src = (const uint32_t*)Ab;
    uint32_t* dst = (uint32_t*)lA;
    for (int t = tid; t < 64 * K / 2; t += 256) dst[t] = src[t];
    if (MODE == 2) {
      const uint32_t* ys = (const uint32_t*)(yres + (size_t)m0 * N);
      uint32_t* yd = (uint32_t*)lOut;
      for (int t = tid; t < 64 * N / 2; t += 256) yd[t] = ys[t];
    }
  }
#endif

  // ---- Stage W transposed: lWt[n*KP + k] = (bf16)W[k*N + n].
  for (int t = tid; t < K * N; t += 256) {
    int kk = t >> NSH, nn = t & (N - 1);
    lWt[nn * KP + kk] = (bf16_t)W[t];
  }
  async_wait0();
  __syncthreads();

  const int lane = tid & 31;
  const int wave = tid >> 5;
  const int wm   = wave & 3;    // M subtile 0..3 (16 rows each)
  const int wn   = wave >> 2;   // N group 0..1
  const int row  = lane & 15;
  const int half = lane >> 4;

  // ---- A fragments for all k-steps, hoisted (2x ds_load_b128 each).
  // Layout: VGPR q<4 -> K = half*8 + 2q{,+1};  q>=4 -> K = 16 + half*8 + ...
  v16bf afrag[KT];
#pragma unroll
  for (int kt = 0; kt < KT; ++kt) {
    const bf16_t* Ar = &lA[(wm * 16 + row) * K + kt * 32 + half * 8];
    v8bf a0 = *(const v8bf*)(Ar);
    v8bf a1 = *(const v8bf*)(Ar + 16);
    afrag[kt] = __builtin_shufflevector(a0, a1, 0, 1, 2, 3, 4, 5, 6, 7,
                                        8, 9, 10, 11, 12, 13, 14, 15);
  }

  // ---- Accumulate all n-tiles owned by this wave.
  v8f accs[NT2];
#pragma unroll
  for (int jj = 0; jj < NT2; ++jj) {
    const int n0 = (wn + 2 * jj) << 4;
    v8f acc = {0.f, 0.f, 0.f, 0.f, 0.f, 0.f, 0.f, 0.f};
#pragma unroll
    for (int kt = 0; kt < KT; ++kt) {
      // B fragment: lane holds col; K = kt*32 + half*8 + 0..7, then +16 (matches A).
      const bf16_t* Bc = &lWt[(n0 + row) * KP + kt * 32 + half * 8];
      v8bf b0 = *(const v8bf*)(Bc);
      v8bf b1 = *(const v8bf*)(Bc + 16);
      v16bf bfrag = __builtin_shufflevector(b0, b1, 0, 1, 2, 3, 4, 5, 6, 7,
                                            8, 9, 10, 11, 12, 13, 14, 15);
      acc = __builtin_amdgcn_wmma_f32_16x16x32_bf16(
          /*neg_a=*/false, afrag[kt], /*neg_b=*/false, bfrag,
          /*c_mod=*/(short)0, acc, /*reuse_a=*/false, /*reuse_b=*/false);
    }
    accs[jj] = acc;
  }

  // ---- Epilogue into LDS out-tile. C layout: VGPR e -> M = e + 8*half, N = lane&15.
  // MODE2 reads its own yres element from lOut then overwrites it (exclusive owner).
#pragma unroll
  for (int jj = 0; jj < NT2; ++jj) {
    const int col = ((wn + 2 * jj) << 4) + row;
    const float bv  = (MODE == 1) ? bias[col] : 0.f;
    const float s1  = (MODE == 2) ? g1[col] * RSQ1PE : 0.f;
    const float o1  = (MODE == 2) ? bt1[col] : 0.f;
    const float s2  = (MODE == 2) ? g2[col] * RSQ1PE : 0.f;
    const float o2  = (MODE == 2) ? bt2[col] : 0.f;
#pragma unroll
    for (int e = 0; e < 8; ++e) {
      const int lrow = wm * 16 + e + half * 8;  // 0..63
      float val = accs[jj][e];
      if (MODE == 1) val += bv;
      if (MODE == 2) {
        float t = val * s1 + o1;
        t = t > 0.f ? t : 0.f;                       // relu(bn1(.))
        float z = (float)lOut[lrow * N + col] + t;   // residual (staged in LDS)
        z = z * s2 + o2;                             // bn2
        val = z > 0.f ? z : NEG_SLOPE * z;           // leaky
      }
      lOut[lrow * N + col] = (bf16_t)val;
    }
  }
  __syncthreads();

  // ---- Bulk store: 64*N bf16 contiguous at out + m0*N -> async b128 stores.
  bf16_t* Ob = out + (size_t)m0 * N;
#if ASYNC_PATH
  {
    const uint32_t lo = lds_off_of(lOut);
    for (int t = tid; t < 64 * N / 8; t += 256)
      async_st_b128(Ob + t * 8, lo + (uint32_t)t * 16u);
    async_wait0();
  }
#else
  {
    const uint32_t* src = (const uint32_t*)lOut;
    uint32_t* dst = (uint32_t*)Ob;
    for (int t = tid; t < 64 * N / 2; t += 256) dst[t] = src[t];
  }
#endif
}

template <int MODE>
static void launch_gemm(const bf16_t* A, const float* W, const float* bias,
                        const bf16_t* yres, const float* g1, const float* bt1,
                        const float* g2, const float* bt2, bf16_t* out,
                        int K, int N, hipStream_t st) {
  dim3 g(2048), b(256);  // 131072 / 64
#define GEMM_CASE(KK, NN)                                                     \
  if (K == KK && N == NN) {                                                   \
    gemm_kernel<MODE, KK, NN><<<g, b, 0, st>>>(A, W, bias, yres, g1, bt1,     \
                                               g2, bt2, out);                 \
    return;                                                                   \
  }
  GEMM_CASE(32, 32) GEMM_CASE(32, 64) GEMM_CASE(64, 32) GEMM_CASE(64, 64)
  GEMM_CASE(64, 128) GEMM_CASE(128, 64) GEMM_CASE(128, 128)
#undef GEMM_CASE
}

// ------------------------------------------------------- local attention ----
// Per (batch, head): 3x3 replicate-padded window attention, 16x16 pixel tiles,
// k/v halo tiles (18x18xD) staged in LDS as f32. grid = (16,16, B*HEADS).
template <int D>
__global__ void __launch_bounds__(256) attn_kernel(
    const bf16_t* __restrict__ q, const bf16_t* __restrict__ k,
    const bf16_t* __restrict__ v, bf16_t* __restrict__ o) {
  constexpr int C = 4 * D;
  __shared__ float lk[18 * 18 * D];
  __shared__ float lv[18 * 18 * D];
  const int b    = blockIdx.z >> 2;
  const int head = blockIdx.z & 3;
  const int x0 = blockIdx.x * 16, y0 = blockIdx.y * 16;
  const int pbase = b << 16;

  for (int t = threadIdx.x; t < 18 * 18 * D; t += 256) {
    int pos = t / D, dd = t % D;           // D constexpr -> shifts
    int ly = pos / 18, lx = pos - ly * 18;
    int gy = y0 + ly - 1; gy = gy < 0 ? 0 : (gy > 255 ? 255 : gy);
    int gx = x0 + lx - 1; gx = gx < 0 ? 0 : (gx > 255 ? 255 : gx);
    size_t gi = (size_t)(pbase + gy * 256 + gx) * C + head * D + dd;
    lk[t] = (float)k[gi];
    lv[t] = (float)v[gi];
  }
  __syncthreads();

  const int tx = threadIdx.x & 15, ty = threadIdx.x >> 4;
  const size_t pi = (size_t)(pbase + (y0 + ty) * 256 + (x0 + tx)) * C + head * D;
  float qr[D];
#pragma unroll
  for (int dd = 0; dd < D; ++dd) qr[dd] = (float)q[pi + dd];

  float s[9];
  float mx = -1e30f;
#pragma unroll
  for (int kk = 0; kk < 9; ++kk) {
    int dy = kk / 3, dx = kk - dy * 3;
    const float* kp = &lk[((ty + dy) * 18 + tx + dx) * D];
    float dot = 0.f;
#pragma unroll
    for (int dd = 0; dd < D; ++dd) dot = fmaf(qr[dd], kp[dd], dot);
    s[kk] = dot;
    mx = fmaxf(mx, dot);
  }
  float sum = 0.f;
#pragma unroll
  for (int kk = 0; kk < 9; ++kk) { s[kk] = __expf(s[kk] - mx); sum += s[kk]; }
  float inv = 1.f / sum;

  float acc[D];
#pragma unroll
  for (int dd = 0; dd < D; ++dd) acc[dd] = 0.f;
#pragma unroll
  for (int kk = 0; kk < 9; ++kk) {
    int dy = kk / 3, dx = kk - dy * 3;
    const float* vp = &lv[((ty + dy) * 18 + tx + dx) * D];
    float a = s[kk] * inv;
#pragma unroll
    for (int dd = 0; dd < D; ++dd) acc[dd] = fmaf(a, vp[dd], acc[dd]);
  }
#pragma unroll
  for (int dd = 0; dd < D; ++dd) o[pi + dd] = (bf16_t)acc[dd];
}

// ------------------------------------------------------------- head ----
// out = sigmoid(act @ w_last + b_last); C=1 so NCHW output == pixel order.
__global__ void head_kernel(const bf16_t* __restrict__ a, const float* __restrict__ w,
                            const float* __restrict__ bias, float* __restrict__ out) {
  int p = blockIdx.x * blockDim.x + threadIdx.x;
  const bf16_t* ap = a + (size_t)p * 32;
  float acc = bias[0];
#pragma unroll
  for (int c = 0; c < 32; ++c) acc = fmaf((float)ap[c], w[c], acc);
  out[p] = 1.f / (1.f + __expf(-acc));
}

// ------------------------------------------------------------- driver ----
extern "C" void kernel_launch(void* const* d_in, const int* in_sizes, int n_in,
                              void* d_out, int out_size, void* d_ws, size_t ws_size,
                              hipStream_t stream) {
  (void)in_sizes; (void)n_in; (void)out_size; (void)ws_size;
  const int M = 2 * 256 * 256;  // 131072 pixel-rows

  // Input order (setup_inputs insertion order, depth-first):
  // 0:x  1..4:conv0_{w,b,g,bt}  5+10*i: block i {w1,b1,wq,wk,wv,wcat,g1,bt1,g2,bt2}
  // 55:last_w  56:last_b
  const float* x    = (const float*)d_in[0];
  const float* c0w  = (const float*)d_in[1];
  const float* c0b  = (const float*)d_in[2];
  const float* c0g  = (const float*)d_in[3];
  const float* c0bt = (const float*)d_in[4];
  const float* lw   = (const float*)d_in[55];
  const float* lb   = (const float*)d_in[56];

  // 5 rotating activation slots (bf16 NHWC), 32 MiB each -> 160 MiB workspace.
  const size_t SLOT = (size_t)M * 128;
  bf16_t* s[5];
  for (int i = 0; i < 5; ++i) s[i] = (bf16_t*)d_ws + SLOT * i;

  conv0_kernel<<<M / 256, 256, 0, stream>>>(x, c0w, c0b, c0g, c0bt, s[0]);

  static const int chans[5][2] = {{32, 32}, {32, 64}, {64, 128}, {128, 64}, {64, 32}};
  int cur = 0;
  for (int bi = 0; bi < 5; ++bi) {
    const float* w1   = (const float*)d_in[5 + 10 * bi + 0];
    const float* b1   = (const float*)d_in[5 + 10 * bi + 1];
    const float* wq   = (const float*)d_in[5 + 10 * bi + 2];
    const float* wk   = (const float*)d_in[5 + 10 * bi + 3];
    const float* wv   = (const float*)d_in[5 + 10 * bi + 4];
    const float* wcat = (const float*)d_in[5 + 10 * bi + 5];
    const float* g1   = (const float*)d_in[5 + 10 * bi + 6];
    const float* bt1  = (const float*)d_in[5 + 10 * bi + 7];
    const float* g2   = (const float*)d_in[5 + 10 * bi + 8];
    const float* bt2  = (const float*)d_in[5 + 10 * bi + 9];
    const int cin = chans[bi][0], cout = chans[bi][1];
    const int iy = (cur + 1) % 5, iq = (cur + 2) % 5,
              ik = (cur + 3) % 5, iv = (cur + 4) % 5;

    // y = x @ w1 + b1
    launch_gemm<1>(s[cur], w1, b1, nullptr, nullptr, nullptr, nullptr, nullptr,
                   s[iy], cin, cout, stream);
    // q, k, v projections (head-major channel order matches reshape)
    launch_gemm<0>(s[iy], wq, nullptr, nullptr, nullptr, nullptr, nullptr, nullptr,
                   s[iq], cout, cout, stream);
    launch_gemm<0>(s[iy], wk, nullptr, nullptr, nullptr, nullptr, nullptr, nullptr,
                   s[ik], cout, cout, stream);
    launch_gemm<0>(s[iy], wv, nullptr, nullptr, nullptr, nullptr, nullptr, nullptr,
                   s[iv], cout, cout, stream);
    // local window attention -> o (reuses slot `cur`)
    dim3 ag(16, 16, 8);
    const int d = cout / 4;
    if (d == 8)       attn_kernel<8><<<ag, 256, 0, stream>>>(s[iq], s[ik], s[iv], s[cur]);
    else if (d == 16) attn_kernel<16><<<ag, 256, 0, stream>>>(s[iq], s[ik], s[iv], s[cur]);
    else              attn_kernel<32><<<ag, 256, 0, stream>>>(s[iq], s[ik], s[iv], s[cur]);
    // z = leaky(bn2(y + relu(bn1(o @ wcat))))   (fully fused epilogue)
    launch_gemm<2>(s[cur], wcat, nullptr, s[iy], g1, bt1, g2, bt2,
                   s[iq], cout, cout, stream);
    cur = iq;
  }

  head_kernel<<<M / 256, 256, 0, stream>>>(s[cur], lw, lb, (float*)d_out);
}
